// LSTMGCN_71004399337892
// MI455X (gfx1250) — matrix operand order repaired and checked by reading, not compiled, not run on hardware
//
#include <hip/hip_runtime.h>
#include <hip/hip_bf16.h>

// ---------------- types for WMMA / TDM ----------------
typedef __bf16 v16bf __attribute__((ext_vector_type(16)));
typedef __bf16 v8bf  __attribute__((ext_vector_type(8)));
typedef float  v8f   __attribute__((ext_vector_type(8)));
typedef unsigned int v4u  __attribute__((ext_vector_type(4)));
typedef int          v8i_ __attribute__((ext_vector_type(8)));
typedef int          v4i_ __attribute__((ext_vector_type(4)));

#define HDIM 128
#define FDIM 6

#if __has_builtin(__builtin_amdgcn_tensor_load_to_lds)
#define HAVE_TDM 1
#endif

// LDS layout for the GEMM kernel (dynamic shared):
//   [0,          40960)  weight panel buffer 0 (512 rows x 80B padded)
//   [40960,      81920)  weight panel buffer 1
//   gates buffer (4x16x132 f32 = 33792B) aliases [0, 33792) after GEMM phase
#define PANEL_STRIDE 80          // 64B row + 16B TDM pad -> 16B aligned, bank-spread
#define PANEL_BYTES  (512 * PANEL_STRIDE)
#define GEMM_LDS_BYTES (2 * PANEL_BYTES)

static __device__ __forceinline__ unsigned short f32_to_bf16(float f) {
    unsigned int u = __float_as_uint(f);
    unsigned int r = (u + 0x7FFFu + ((u >> 16) & 1u)) >> 16;   // RNE
    return (unsigned short)r;
}

static __device__ __forceinline__ float sigmoidf_(float x) {
    return 1.0f / (1.0f + __expf(-x));
}

// ---------------- CSR construction ----------------
__global__ void zero_i32_kernel(int* p, int n) {
    int i = blockIdx.x * blockDim.x + threadIdx.x;
    if (i < n) p[i] = 0;
}

__global__ void count_deg_kernel(const int* __restrict__ dst, int* deg, int E) {
    int e = blockIdx.x * blockDim.x + threadIdx.x;
    if (e < E) atomicAdd(&deg[dst[e]], 1);
}

__global__ __launch_bounds__(1024) void scan_kernel(const int* __restrict__ deg,
                                                    int* __restrict__ offs, int n) {
    __shared__ int tmp[1024];
    __shared__ int carry;
    int t = threadIdx.x;
    if (t == 0) carry = 0;
    __syncthreads();
    for (int base = 0; base < n; base += 1024) {
        int v = (base + t < n) ? deg[base + t] : 0;
        tmp[t] = v;
        __syncthreads();
        for (int off = 1; off < 1024; off <<= 1) {
            int add = (t >= off) ? tmp[t - off] : 0;
            __syncthreads();
            tmp[t] += add;
            __syncthreads();
        }
        if (base + t < n) offs[base + t] = carry + tmp[t] - v;  // exclusive
        __syncthreads();
        if (t == 0) carry += tmp[1023];
        __syncthreads();
    }
}

__global__ void fill_csr_kernel(const int* __restrict__ src, const int* __restrict__ dst,
                                const int* __restrict__ offs, int* fill, int* csr, int E) {
    int e = blockIdx.x * blockDim.x + threadIdx.x;
    if (e >= E) return;
    int d = dst[e];
    int pos = offs[d] + atomicAdd(&fill[d], 1);
    csr[pos] = src[e];
}

// canonicalize per-node edge lists so float-sum order is deterministic
__global__ void sort_csr_kernel(const int* __restrict__ offs, const int* __restrict__ deg,
                                int* csr, int N) {
    int n = blockIdx.x * blockDim.x + threadIdx.x;
    if (n >= N) return;
    int b = offs[n], d = deg[n];
    for (int i = 1; i < d; i++) {
        int key = csr[b + i];
        int j = i - 1;
        while (j >= 0 && csr[b + j] > key) { csr[b + j + 1] = csr[b + j]; j--; }
        csr[b + j + 1] = key;
    }
}

// ---------------- weight / state prep ----------------
__global__ void prepw_kernel(const float* __restrict__ W_ih, const float* __restrict__ W_hh,
                             const float* __restrict__ b_ih, const float* __restrict__ b_hh,
                             unsigned short* __restrict__ Wcat, float* __restrict__ bcat) {
    int idx = blockIdx.x * blockDim.x + threadIdx.x;
    if (idx < 512 * 256) {
        int j = idx >> 8, k = idx & 255;
        float v = (k < HDIM) ? W_ih[j * HDIM + k] : W_hh[j * HDIM + (k - HDIM)];
        Wcat[idx] = f32_to_bf16(v);
    }
    if (idx < 512) bcat[idx] = b_ih[idx] + b_hh[idx];
}

__global__ void initstate_kernel(float* h, float* c, unsigned short* Acat, int N) {
    int idx = blockIdx.x * blockDim.x + threadIdx.x;
    if (idx >= N * HDIM) return;
    h[idx] = 0.0f;
    c[idx] = 0.0f;
    int n = idx >> 7, col = idx & 127;
    Acat[(size_t)n * 256 + 128 + col] = 0;   // bf16 zero for h-half of A operand
}

// ---------------- input GCN layer ----------------
__global__ void aggF_kernel(const float* __restrict__ feat, const int* __restrict__ offs,
                            const int* __restrict__ deg, const int* __restrict__ csr,
                            float* __restrict__ aggF, int N) {
    int n = blockIdx.x * blockDim.x + threadIdx.x;
    if (n >= N) return;
    float s[FDIM];
#pragma unroll
    for (int k = 0; k < FDIM; k++) s[k] = 0.0f;
    int b = offs[n], d = deg[n];
    for (int e = 0; e < d; e++) {
        int si = csr[b + e];
#pragma unroll
        for (int k = 0; k < FDIM; k++) s[k] += feat[si * FDIM + k];
    }
#pragma unroll
    for (int k = 0; k < FDIM; k++) aggF[n * FDIM + k] = s[k];
}

__global__ void inlayer_kernel(const float* __restrict__ aggF, const float* __restrict__ W_in,
                               const float* __restrict__ b_in, float* __restrict__ x, int N) {
    int idx = blockIdx.x * blockDim.x + threadIdx.x;
    if (idx >= N * HDIM) return;
    int n = idx >> 7, col = idx & 127;
    float s = b_in[col];
#pragma unroll
    for (int k = 0; k < FDIM; k++) s += aggF[n * FDIM + k] * W_in[k * HDIM + col];
    x[idx] = s > 0.0f ? s : 0.0f;
}

// ---------------- per-step aggregation: Acat[:,0:128] = bf16(segment_sum(inp)) -----
__global__ __launch_bounds__(128) void agg_kernel(const float* __restrict__ hin,
                                                  const int* __restrict__ offs,
                                                  const int* __restrict__ deg,
                                                  const int* __restrict__ csr,
                                                  unsigned short* __restrict__ Acat, int N) {
    int n = blockIdx.x;
    int col = threadIdx.x;
    int b = offs[n], d = deg[n];
    float s = 0.0f;
    for (int e = 0; e < d; e++) {
        int si = csr[b + e];
        s += hin[(size_t)si * HDIM + col];
    }
    Acat[(size_t)n * 256 + col] = f32_to_bf16(s);
}

#ifdef HAVE_TDM
// Issue one TDM descriptor: copy the 512x32 bf16 K-panel (kk) of Wcat into LDS,
// padded 4 dwords every 16 dwords -> 80B LDS row stride (aligned + bank-spread).
static __device__ __forceinline__ void tdm_load_panel(const unsigned short* Wcat, int kk,
                                                      unsigned lds_off) {
    unsigned long long ga = (unsigned long long)(uintptr_t)Wcat + (unsigned)kk * 64u;
    v4u g0;
    g0[0] = 1u;                                   // count=1, user descriptor
    g0[1] = lds_off;                              // LDS byte address
    g0[2] = (unsigned)(ga & 0xFFFFFFFFull);       // global_addr[31:0]
    g0[3] = ((unsigned)(ga >> 32) & 0x01FFFFFFu)  // global_addr[56:32]
            | 0x80000000u;                        // type=2 ("image")
    v8i_ g1;
    g1[0] = (int)((1u << 16)     // data_size = 2B
                | (1u << 20)     // pad_enable
                | (3u << 22)     // pad_interval: every 16 dwords (64B = one row)
                | (3u << 25));   // pad_amount: 4 dwords (16B) -> 80B stride
    g1[1] = (int)(256u << 16);   // tensor_dim0 = 256 (low16 at bit48)
    g1[2] = (int)(512u << 16);   // tensor_dim0 hi=0 | tensor_dim1 = 512 (low16 at bit80)
    g1[3] = (int)(32u << 16);    // tensor_dim1 hi=0 | tile_dim0 = 32
    g1[4] = 512;                 // tile_dim1 = 512, tile_dim2 = 0
    g1[5] = 256;                 // tensor_dim0_stride = 256 elements
    g1[6] = 0;
    g1[7] = 0;
    v4i_ z4 = {0, 0, 0, 0};      // groups 2/3 unused (2-D tile)
    v8i_ z8 = {0, 0, 0, 0, 0, 0, 0, 0};
    // 6-arg form (clang-23 / therock headers): g0, g1, g2, g3, extra group, cpol
    __builtin_amdgcn_tensor_load_to_lds(g0, g1, z4, z4, z8, 0);
}
#endif

// ---------------- fused GEMM (WMMA bf16) + LSTM cell ----------------
// gates = [a|h] @ [W_ih|W_hh]^T + (b_ih+b_hh)
// 32-node tile per block; wave w (0..3) computes gate w's 32x128 pre-activation.
// Weight K-panels are staged into LDS by the Tensor Data Mover, double-buffered
// across the K loop (TENSORcnt), so WMMAs consume low-latency ds_load fragments
// while the DMA for the next panel runs underneath.
__global__ __launch_bounds__(128) void gemm_lstm_kernel(unsigned short* __restrict__ Acat,
                                                        const unsigned short* __restrict__ Wcat,
                                                        const float* __restrict__ bcat,
                                                        float* __restrict__ h,
                                                        float* __restrict__ c, int N) {
    extern __shared__ char smem[];         // 2 x PANEL_BYTES; gates buffer aliases after
    const int tid = threadIdx.x;
    const int wave = tid >> 5;             // gate index 0..3
    const int lane = tid & 31;
    const int laneHalf = lane >> 4;        // 0/1
    const int lanem = lane & 15;
    const int tileM = blockIdx.x;          // 32 rows per block

    // accumulators pre-loaded with bias (every M in a lane shares column j)
    v8f acc0[8], acc1[8];
#pragma unroll
    for (int jt = 0; jt < 8; jt++) {
        float bv = bcat[wave * HDIM + jt * 16 + lanem];
        v8f a0;
#pragma unroll
        for (int r = 0; r < 8; r++) a0[r] = bv;
        acc0[jt] = a0;
        acc1[jt] = a0;
    }

    int r0 = tileM * 32 + lanem;
    int r1 = tileM * 32 + 16 + lanem;
    if (r0 >= N) r0 = N - 1;               // safety clamp (results discarded below)
    if (r1 >= N) r1 = N - 1;
    const unsigned short* arow0 = Acat + (size_t)r0 * 256;
    const unsigned short* arow1 = Acat + (size_t)r1 * 256;

    // A fragment: lane half picks K interleave 0-7/16-23 vs 8-15/24-31
    auto loadA = [&](const unsigned short* arow, int kk) -> v16bf {
        int k0 = kk * 32 + laneHalf * 8;
        v8bf lo = *reinterpret_cast<const v8bf*>(arow + k0);
        v8bf hi = *reinterpret_cast<const v8bf*>(arow + k0 + 16);
        v16bf a;
#pragma unroll
        for (int q = 0; q < 8; q++) { a[q] = lo[q]; a[q + 8] = hi[q]; }
        return a;
    };

#ifndef HAVE_TDM
    // fallback: cooperative global->LDS copy of one K-panel (same LDS layout)
    auto stage_panel = [&](char* buf, int kk) {
#pragma unroll
        for (int r = 0; r < 4; r++) {
            int j = tid * 4 + r;           // 0..511
            const uint4* g = reinterpret_cast<const uint4*>(Wcat + (size_t)j * 256 + kk * 32);
            uint4 d0 = g[0];
            uint4 d1 = g[1];
            char* rowp = buf + j * PANEL_STRIDE;
            *reinterpret_cast<uint4*>(rowp) = d0;
            *reinterpret_cast<uint4*>(rowp + 16) = d1;
        }
    };
#endif

    // prologue: stage panel 0
#ifdef HAVE_TDM
    if (wave == 0) {
        tdm_load_panel(Wcat, 0, (unsigned)(uintptr_t)smem);
        __builtin_amdgcn_s_wait_tensorcnt(0);
    }
#else
    stage_panel(smem, 0);
#endif
    __syncthreads();

    for (int kk = 0; kk < 8; kk++) {       // K = 256 in panels of 32
        char* bufc = smem + (size_t)(kk & 1) * PANEL_BYTES;
#ifdef HAVE_TDM
        if (wave == 0 && kk < 7) {
            tdm_load_panel(Wcat, kk + 1,
                           (unsigned)(uintptr_t)(smem + (size_t)((kk + 1) & 1) * PANEL_BYTES));
        }
#else
        if (kk < 7) stage_panel(smem + (size_t)((kk + 1) & 1) * PANEL_BYTES, kk + 1);
#endif
        v16bf a0 = loadA(arow0, kk);
        v16bf a1 = loadA(arow1, kk);
#pragma unroll
        for (int jt = 0; jt < 8; jt++) {
            int j = wave * HDIM + jt * 16 + lanem;
            const char* rowp = bufc + j * PANEL_STRIDE + laneHalf * 16;
            v8bf blo = *reinterpret_cast<const v8bf*>(rowp);
            v8bf bhi = *reinterpret_cast<const v8bf*>(rowp + 32);
            v16bf bfrag;
#pragma unroll
            for (int q = 0; q < 8; q++) { bfrag[q] = blo[q]; bfrag[q + 8] = bhi[q]; }
            acc0[jt] = __builtin_amdgcn_wmma_f32_16x16x32_bf16(
                false, a0, false, bfrag, (short)0, acc0[jt], false, false);
            acc1[jt] = __builtin_amdgcn_wmma_f32_16x16x32_bf16(
                false, a1, false, bfrag, (short)0, acc1[jt], false, false);
        }
#ifdef HAVE_TDM
        if (wave == 0 && kk < 7) __builtin_amdgcn_s_wait_tensorcnt(0);
#endif
        __syncthreads();                   // publish next panel / retire this one
    }

    // gates exchange buffer aliases the (now dead) weight panels
    float* gbuf = reinterpret_cast<float*>(smem);   // [4][16][HDIM+4]
    auto gref = [&](int g, int m, int col) -> float& {
        return gbuf[(g * 16 + m) * (HDIM + 4) + col];
    };

    // two passes of 16 rows through the LDS gate buffer
#pragma unroll
    for (int half = 0; half < 2; half++) {
        // scatter D layout (lane: N=lanem, M = r + 8*laneHalf) into LDS
#pragma unroll
        for (int jt = 0; jt < 8; jt++) {
#pragma unroll
            for (int r = 0; r < 8; r++) {
                float v = (half == 0) ? acc0[jt][r] : acc1[jt][r];
                gref(wave, laneHalf * 8 + r, jt * 16 + lanem) = v;
            }
        }
        __syncthreads();

        // LSTM cell update: thread owns column `tid` for 16 nodes of this half
        const int col = tid;
#pragma unroll 4
        for (int m = 0; m < 16; m++) {
            int node = tileM * 32 + half * 16 + m;
            if (node >= N) break;
            float si = sigmoidf_(gref(0, m, col));
            float sf = sigmoidf_(gref(1, m, col));
            float tg = tanhf(gref(2, m, col));
            float so = sigmoidf_(gref(3, m, col));
            size_t idx = (size_t)node * HDIM + col;
            float cn = sf * c[idx] + si * tg;
            float hn = so * tanhf(cn);
            c[idx] = cn;
            h[idx] = hn;
            Acat[(size_t)node * 256 + 128 + col] = f32_to_bf16(hn);  // h-half for next step
        }
        __syncthreads();   // LDS reused by next half
    }
}

// ---------------- output GCN layer ----------------
__global__ __launch_bounds__(128) void proj_kernel(const float* __restrict__ h,
                                                   const float* __restrict__ W_out,
                                                   float* __restrict__ p, int N) {
    int wave = threadIdx.x >> 5, lane = threadIdx.x & 31;
    int n = blockIdx.x * 4 + wave;
    if (n >= N) return;
    float s = 0.0f;
#pragma unroll
    for (int r = 0; r < 4; r++) {
        int col = lane + r * 32;
        s += h[(size_t)n * HDIM + col] * W_out[col];
    }
#pragma unroll
    for (int o = 16; o > 0; o >>= 1) s += __shfl_down(s, o, 32);
    if (lane == 0) p[n] = s;
}

__global__ void out_kernel(const float* __restrict__ p, const int* __restrict__ offs,
                           const int* __restrict__ deg, const int* __restrict__ csr,
                           const float* __restrict__ b_out, float* __restrict__ out, int N) {
    int n = blockIdx.x * blockDim.x + threadIdx.x;
    if (n >= N) return;
    float s = b_out[0];
    int b = offs[n], d = deg[n];
    for (int e = 0; e < d; e++) s += p[csr[b + e]];
    out[n] = s;
}

// ---------------- host orchestration ----------------
extern "C" void kernel_launch(void* const* d_in, const int* in_sizes, int n_in,
                              void* d_out, int out_size, void* d_ws, size_t ws_size,
                              hipStream_t stream) {
    const float* features = (const float*)d_in[0];
    const int*   srcI     = (const int*)d_in[1];
    const int*   dstI     = (const int*)d_in[2];
    const float* W_in     = (const float*)d_in[3];
    const float* b_in     = (const float*)d_in[4];
    const float* W_ih     = (const float*)d_in[5];
    const float* W_hh     = (const float*)d_in[6];
    const float* b_ih     = (const float*)d_in[7];
    const float* b_hh     = (const float*)d_in[8];
    const float* W_out    = (const float*)d_in[9];
    const float* b_out    = (const float*)d_in[10];
    float* out = (float*)d_out;

    const int N = in_sizes[0] / FDIM;
    const int E = in_sizes[1];
    const int DEPTH = 50;

    // carve workspace (512B-aligned chunks)
    char* base = (char*)d_ws;
    size_t off = 0;
    auto carve = [&](size_t bytes) -> char* {
        char* ptr = base + off;
        off = (off + bytes + 511) & ~(size_t)511;
        return ptr;
    };
    unsigned short* Acat = (unsigned short*)carve((size_t)N * 256 * sizeof(unsigned short));
    float* hbuf = (float*)carve((size_t)N * HDIM * sizeof(float));
    float* cbuf = (float*)carve((size_t)N * HDIM * sizeof(float));
    float* xbuf = (float*)carve((size_t)N * HDIM * sizeof(float));
    float* aggF = (float*)carve((size_t)N * FDIM * sizeof(float));
    unsigned short* Wcat = (unsigned short*)carve((size_t)512 * 256 * sizeof(unsigned short));
    float* bcat = (float*)carve(512 * sizeof(float));
    int* deg  = (int*)carve((size_t)N * sizeof(int));
    int* fill = (int*)carve((size_t)N * sizeof(int));
    int* offs = (int*)carve((size_t)N * sizeof(int));
    int* csr  = (int*)carve((size_t)E * sizeof(int));
    float* pbuf = (float*)carve((size_t)N * sizeof(float));
    (void)ws_size; (void)n_in; (void)out_size;

    // --- CSR build (deterministic: per-node lists sorted) ---
    zero_i32_kernel<<<(N + 255) / 256, 256, 0, stream>>>(deg, N);
    zero_i32_kernel<<<(N + 255) / 256, 256, 0, stream>>>(fill, N);
    count_deg_kernel<<<(E + 255) / 256, 256, 0, stream>>>(dstI, deg, E);
    scan_kernel<<<1, 1024, 0, stream>>>(deg, offs, N);
    fill_csr_kernel<<<(E + 255) / 256, 256, 0, stream>>>(srcI, dstI, offs, fill, csr, E);
    sort_csr_kernel<<<(N + 255) / 256, 256, 0, stream>>>(offs, deg, csr, N);

    // --- weight conversion & state init ---
    prepw_kernel<<<(512 * 256 + 255) / 256, 256, 0, stream>>>(W_ih, W_hh, b_ih, b_hh, Wcat, bcat);
    initstate_kernel<<<(N * HDIM + 255) / 256, 256, 0, stream>>>(hbuf, cbuf, Acat, N);

    // --- input GCN layer: x = relu(agg(features) @ W_in + b_in) ---
    aggF_kernel<<<(N + 127) / 128, 128, 0, stream>>>(features, offs, deg, csr, aggF, N);
    inlayer_kernel<<<(N * HDIM + 255) / 256, 256, 0, stream>>>(aggF, W_in, b_in, xbuf, N);

    // --- 1 + DEPTH LSTM-GCN steps ---
    const int numTiles = (N + 31) / 32;
    const float* inp = xbuf;
    for (int t = 0; t < 1 + DEPTH; t++) {
        agg_kernel<<<N, 128, 0, stream>>>(inp, offs, deg, csr, Acat, N);
        gemm_lstm_kernel<<<numTiles, 128, GEMM_LDS_BYTES, stream>>>(Acat, Wcat, bcat,
                                                                    hbuf, cbuf, N);
        inp = hbuf;
    }

    // --- output GCN layer: out = agg(h) @ W_out + b_out ---
    proj_kernel<<<(N + 3) / 4, 128, 0, stream>>>(hbuf, W_out, pbuf, N);
    out_kernel<<<(N + 255) / 256, 256, 0, stream>>>(pbuf, offs, deg, csr, b_out, out, N);
}
